// LocalEdgeEncoder_65824668779103
// MI455X (gfx1250) — matrix operand order repaired and checked by reading, not compiled, not run on hardware
//
#include <hip/hip_runtime.h>

// LocalEdgeEncoder for MI455X (gfx1250, wave32).
// Pass 1: scatter node sums/degrees with f32 atomics (node_sum = 25.6MB -> L2 resident).
// Pass 2: per 16-edge tile, build exclusive-mean aggregates directly in the WMMA
//         16-bit A-matrix register layout and run D = A*W + C with
//         v_wmma_f32_16x16x32_f16 (K=64 -> 2 chained WMMAs per 16x16 tile),
//         then fuse bias + gate-projection (wave shuffle reduce) + sigmoid blend.

#define N_NODES   100000
#define E_EDGES   1000000
#define DIM       64
#define NTILES    (E_EDGES / 16)   // 62500

typedef __attribute__((ext_vector_type(16))) _Float16 v16h;
typedef __attribute__((ext_vector_type(8)))  float    v8f;

// ---------------------------------------------------------------------------
// Pass 1: node_sum / deg scatter. One thread per (edge, channel).
// ---------------------------------------------------------------------------
__global__ void lee_scatter(const float* __restrict__ emb,
                            const int*   __restrict__ eidx,
                            float*       __restrict__ node_sum,
                            float*       __restrict__ deg) {
    long long tid = (long long)blockIdx.x * blockDim.x + threadIdx.x;
    if (tid >= (long long)E_EDGES * DIM) return;
    int e  = (int)(tid >> 6);
    int ch = (int)(tid & 63);
    float x = emb[tid];
    int u = eidx[2 * e];
    int v = eidx[2 * e + 1];
    atomicAdd(&node_sum[(long long)u * DIM + ch], x);
    atomicAdd(&node_sum[(long long)v * DIM + ch], x);
    if (ch == 0) {
        atomicAdd(&deg[u], 1.0f);
        atomicAdd(&deg[v], 1.0f);
    }
}

// ---------------------------------------------------------------------------
// Prep: swizzle Wf/Wb (64x64 f32, row-major [k][n]) into f16 B-matrix register
// layout. Per N-tile nt and K-part (K 0..31 / 32..63):
//   lane: n = nt*16 + (lane&15), kgroup = lane>>4
//   v16h element 2j/2j+1 holds K = part*32 + (j>>2)*16 + kgroup*8 + 2*(j&3) (+1)
// Stored as Bswz[(((mat*4+nt)*2+part)*32 + lane)*16 + elem].
// ---------------------------------------------------------------------------
__global__ void lee_prep_weights(const float* __restrict__ Wf,
                                 const float* __restrict__ Wb,
                                 _Float16*    __restrict__ Bswz) {
    int w = blockIdx.x * blockDim.x + threadIdx.x;
    if (w >= 512) return;                 // 2 mats * 4 nt * 2 parts * 32 lanes
    int lane = w & 31;
    int part = (w >> 5) & 1;
    int nt   = (w >> 6) & 3;
    int mat  = (w >> 8) & 1;
    const float* W = mat ? Wb : Wf;
    int n  = nt * 16 + (lane & 15);
    int kg = lane >> 4;
    _Float16* dst = Bswz + (size_t)((((mat * 4 + nt) * 2 + part) * 32) + lane) * 16;
#pragma unroll
    for (int j = 0; j < 8; ++j) {
        int k = part * 32 + (j >> 2) * 16 + kg * 8 + 2 * (j & 3);
        dst[2 * j]     = (_Float16)W[k * DIM + n];
        dst[2 * j + 1] = (_Float16)W[(k + 1) * DIM + n];
    }
}

// ---------------------------------------------------------------------------
// Pass 2: fused aggregate + dual GEMM (WMMA) + gate + blend.
// One wave (32 lanes) per tile of 16 edges; 256-thread blocks = 8 tiles/block.
// ---------------------------------------------------------------------------
__global__ void __launch_bounds__(256)
lee_edge_wmma(const float*    __restrict__ emb,
              const int*      __restrict__ eidx,
              const float*    __restrict__ node_sum,
              const float*    __restrict__ deg,
              const _Float16* __restrict__ Bswz,
              const float*    __restrict__ bf,
              const float*    __restrict__ bb,
              const float*    __restrict__ Wg,
              const float*    __restrict__ bg,
              float*          __restrict__ out) {
    int wave = blockIdx.x * (blockDim.x >> 5) + (threadIdx.x >> 5);
    if (wave >= NTILES) return;           // wave-uniform: EXEC all-1s inside

    int lane = threadIdx.x & 31;
    int m    = lane & 15;                 // row within 16-edge tile / N within N-tile
    int hh   = lane >> 4;                 // lane-half selector
    int e    = wave * 16 + m;

    int u = eidx[2 * e];
    int v = eidx[2 * e + 1];
    float du = deg[u];
    float dv = deg[v];
    float inv_u = (du > 1.0f) ? 1.0f / fmaxf(du - 1.0f, 1.0f) : 0.0f;
    float inv_v = (dv > 1.0f) ? 1.0f / fmaxf(dv - 1.0f, 1.0f) : 0.0f;

    const float* er  = emb      + (long long)e * DIM;
    const float* nsu = node_sum + (long long)u * DIM;
    const float* nsv = node_sum + (long long)v * DIM;

    // Build A fragments (16-bit A 16x32 layout, ISA 7.12.2):
    //   lane half hh, VGPR j(0..3): K = 8*hh + 2j,2j+1 ; j(4..7): K = 16+8*hh+...
    //   => v16h element s*8+t holds K = part*32 + s*16 + 8*hh + t
    v16h Au_lo{}, Au_hi{}, Av_lo{}, Av_hi{};
#pragma unroll
    for (int p = 0; p < 2; ++p) {
#pragma unroll
        for (int s = 0; s < 2; ++s) {
            int base = p * 32 + s * 16 + 8 * hh;
#pragma unroll
            for (int t = 0; t < 8; ++t) {
                int k = base + t;
                float ev = er[k];
                _Float16 au = (_Float16)((nsu[k] - ev) * inv_u);
                _Float16 av = (_Float16)((nsv[k] - ev) * inv_v);
                int idx = s * 8 + t;
                if (p == 0) { Au_lo[idx] = au; Av_lo[idx] = av; }
                else        { Au_hi[idx] = au; Av_hi[idx] = av; }
            }
        }
    }

    // Dual GEMM over 4 N-tiles, K=64 via two chained 16x16x32 WMMAs each.
    const v16h* bptr = (const v16h*)Bswz;
    v8f accF[4], accB[4];
#pragma unroll
    for (int nt = 0; nt < 4; ++nt) {
        v16h Bf_lo = bptr[((0 * 4 + nt) * 2 + 0) * 32 + lane];
        v16h Bf_hi = bptr[((0 * 4 + nt) * 2 + 1) * 32 + lane];
        v16h Bb_lo = bptr[((1 * 4 + nt) * 2 + 0) * 32 + lane];
        v16h Bb_hi = bptr[((1 * 4 + nt) * 2 + 1) * 32 + lane];
        v8f c{};
        c = __builtin_amdgcn_wmma_f32_16x16x32_f16(false, Au_lo, false, Bf_lo,
                                                   (short)0, c, false, false);
        c = __builtin_amdgcn_wmma_f32_16x16x32_f16(false, Au_hi, false, Bf_hi,
                                                   (short)0, c, false, false);
        accF[nt] = c;
        v8f d{};
        d = __builtin_amdgcn_wmma_f32_16x16x32_f16(false, Av_lo, false, Bb_lo,
                                                   (short)0, d, false, false);
        d = __builtin_amdgcn_wmma_f32_16x16x32_f16(false, Av_hi, false, Bb_hi,
                                                   (short)0, d, false, false);
        accB[nt] = d;
    }

    // Bias + gate projection. C/D layout: lane col N = nt*16+m,
    // VGPR r -> row M = r + 8*hh.
    float wgv[4], bfv[4], bbv[4];
#pragma unroll
    for (int nt = 0; nt < 4; ++nt) {
        bfv[nt] = bf[nt * 16 + m];
        bbv[nt] = bb[nt * 16 + m];
        wgv[nt] = Wg[nt * 16 + m];
    }
    float bgv = bg[0];

    float p[8];
#pragma unroll
    for (int r = 0; r < 8; ++r) p[r] = 0.0f;
#pragma unroll
    for (int nt = 0; nt < 4; ++nt) {
#pragma unroll
        for (int r = 0; r < 8; ++r) {
            float F = accF[nt][r] + bfv[nt];
            float B = accB[nt][r] + bbv[nt];
            accF[nt][r] = F;
            accB[nt][r] = B;
            p[r] += (F + B) * wgv[nt];
        }
    }

    // Row reduction across the 16 lanes holding that row's 16 columns
    // (xor masks 1,2,4,8 stay within each 16-lane half under wave32).
    float gate[8];
#pragma unroll
    for (int r = 0; r < 8; ++r) {
        float s = p[r];
        s += __shfl_xor(s, 1, 32);
        s += __shfl_xor(s, 2, 32);
        s += __shfl_xor(s, 4, 32);
        s += __shfl_xor(s, 8, 32);
        gate[r] = 1.0f / (1.0f + __expf(-(s + bgv)));
    }

    // Blend + store: out[row][n], row = wave*16 + r + 8*hh, n = nt*16 + m.
#pragma unroll
    for (int nt = 0; nt < 4; ++nt) {
#pragma unroll
        for (int r = 0; r < 8; ++r) {
            long long row = (long long)wave * 16 + r + 8 * hh;
            out[row * DIM + nt * 16 + m] =
                gate[r] * accF[nt][r] + (1.0f - gate[r]) * accB[nt][r];
        }
    }
}

// ---------------------------------------------------------------------------
extern "C" void kernel_launch(void* const* d_in, const int* in_sizes, int n_in,
                              void* d_out, int out_size, void* d_ws, size_t ws_size,
                              hipStream_t stream) {
    const float* emb = (const float*)d_in[0];
    const float* Wf  = (const float*)d_in[1];
    const float* bf  = (const float*)d_in[2];
    const float* Wb  = (const float*)d_in[3];
    const float* bb  = (const float*)d_in[4];
    const float* Wg  = (const float*)d_in[5];
    const float* bg  = (const float*)d_in[6];
    const int*  eidx = (const int*)d_in[7];   // int32 (jax canonicalized)
    float* out = (float*)d_out;

    // Workspace layout (needs ~26.02 MB):
    //   [0)                    node_sum : N_NODES*64 f32  (25,600,000 B)
    //   [25,600,000)           deg      : N_NODES   f32   (   400,000 B)
    //   [26,000,000)           Bswz     : 2*4*2*32*16 f16 (    16,384 B), 32B aligned
    char* ws = (char*)d_ws;
    float*    node_sum = (float*)ws;
    float*    deg      = (float*)(ws + (size_t)N_NODES * DIM * sizeof(float));
    _Float16* Bswz     = (_Float16*)(ws + (size_t)N_NODES * DIM * sizeof(float)
                                        + (size_t)N_NODES * sizeof(float));

    size_t zero_bytes = (size_t)N_NODES * DIM * sizeof(float)
                      + (size_t)N_NODES * sizeof(float);
    hipMemsetAsync(d_ws, 0, zero_bytes, stream);

    lee_prep_weights<<<2, 256, 0, stream>>>(Wf, Wb, Bswz);

    long long scatter_items = (long long)E_EDGES * DIM;        // 64M
    int scatter_blocks = (int)((scatter_items + 255) / 256);   // 250,000
    lee_scatter<<<scatter_blocks, 256, 0, stream>>>(emb, eidx, node_sum, deg);

    int waves_per_block = 8;                                   // 256 threads
    int blocks = (NTILES + waves_per_block - 1) / waves_per_block;  // 7813
    lee_edge_wmma<<<blocks, 256, 0, stream>>>(emb, eidx, node_sum, deg,
                                              Bswz, bf, bb, Wg, bg, out);
}
